// LSTM_42571715838591
// MI455X (gfx1250) — matrix operand order repaired and checked by reading, not compiled, not run on hardware
//
#include <hip/hip_runtime.h>
#include <hip/hip_bf16.h>

typedef __attribute__((ext_vector_type(16))) __bf16 v16bf;
typedef __attribute__((ext_vector_type(8)))  float  v8f;

#define T_STEPS 512
#define BATCH   64
#define HID     512
#define KIN0    256
#define WG_PER_LAYER 16
#define HSLICE  32        // hidden units owned per WG
#define NCOLS   128       // 4 gates * HSLICE
#define CHUNK   64        // final output chunk length
#define NWG_LSTM (3 * WG_PER_LAYER)

// ---- LDS layout (persistent kernel), bytes ----
#define LDS_WHH_OFF   0            // [NCOLS][512] bf16   (128 KB, resident all steps)
#define LDS_XST_OFF   131072       // x_t stage [64][<=512] bf16 (64 KB)
#define LDS_HST_OFF   196608       // h_{t-1} stage [64][512] bf16 (64 KB)
#define LDS_GBUF_OFF  262144       // [64][128] f32       (32 KB)
#define LDS_CBUF_OFF  294912       // [64][32]  f32       (8 KB)
#define LDS_BIAS_OFF  303104       // [128]     f32       (512 B)
#define LDS_TOTAL     303616

// ---- workspace layout, bytes ----
#define WS_XBF_OFF    0                       // 512*64*256 bf16 = 16 MB
#define WS_FCW_OFF    16777216                // 128*512 bf16
#define WS_HBUF_OFF   16908288                // [3][2][64][512] bf16
#define WS_HS_OFF     17301504                // [64][64][512] bf16
#define WS_WIH0_OFF   21495808                // 2048*256 bf16 = 1 MB
#define WS_WIH1_OFF   22544384                // 2048*512 bf16 = 2 MB
#define WS_WIH2_OFF   24641536                // 2048*512 bf16 = 2 MB
#define WS_BAR_OFF    26738688                // barrier counter

__device__ __forceinline__ unsigned short f2bf(float f) {
  union { float f; unsigned u; } v; v.f = f;
  unsigned r = v.u + 0x7FFFu + ((v.u >> 16) & 1u);   // round-to-nearest-even
  return (unsigned short)(r >> 16);
}

union FragU { v16bf v; uint4 q[2]; };

// A fragment, 16x32 bf16, row-major source with leading dim ld (halves).
// Lane L<16: row L, K = {0..7, 16..23}; lane L>=16: row L-16, K = {8..15, 24..31}.
__device__ __forceinline__ v16bf load_a_bf16(const unsigned short* tile, int ld, int lane) {
  const unsigned short* p = tile + (size_t)(lane & 15) * ld + ((lane >> 4) << 3);
  FragU u;
  u.q[0] = *(const uint4*)(p);
  u.q[1] = *(const uint4*)(p + 16);
  return u.v;
}

// B fragment, 32x16 bf16, weights stored [col][k] with leading dim ld (halves).
// Lane L<16: col L, K = {0..7, 16..23}; lane L>=16: col L-16, K = {8..15, 24..31}.
__device__ __forceinline__ v16bf load_b_bf16(const unsigned short* tile, int ld, int lane) {
  const unsigned short* p = tile + (size_t)(lane & 15) * ld + ((lane >> 4) << 3);
  FragU u;
  u.q[0] = *(const uint4*)(p);
  u.q[1] = *(const uint4*)(p + 16);
  return u.v;
}

// CDNA5 async global->LDS copy, 16B per lane. Tracked by ASYNCcnt.
__device__ __forceinline__ void async_copy_b128(unsigned lds_byte_off, const void* gaddr) {
  asm volatile("global_load_async_to_lds_b128 %0, %1, off"
               :: "v"(lds_byte_off), "v"(gaddr) : "memory");
}
__device__ __forceinline__ void wait_asynccnt0() {
  asm volatile("s_wait_asynccnt 0x0" ::: "memory");
}

__device__ __forceinline__ float sigmoidf_(float x) { return 1.0f / (1.0f + __expf(-x)); }

// ---------------- prep: bf16 conversions + state init ----------------
extern "C" __global__ void lstm_prep_kernel(const float* __restrict__ x,
                                            const float* __restrict__ fcW,
                                            const float* __restrict__ Wih0,
                                            const float* __restrict__ Wih1,
                                            const float* __restrict__ Wih2,
                                            unsigned short* __restrict__ xbf,
                                            unsigned short* __restrict__ fcWbf,
                                            unsigned short* __restrict__ wih0bf,
                                            unsigned short* __restrict__ wih1bf,
                                            unsigned short* __restrict__ wih2bf,
                                            unsigned short* __restrict__ hbuf,
                                            unsigned* __restrict__ barrier_cnt) {
  size_t tid = (size_t)blockIdx.x * blockDim.x + threadIdx.x;
  size_t stride = (size_t)gridDim.x * blockDim.x;
  const size_t nx = (size_t)T_STEPS * BATCH * KIN0;
  for (size_t i = tid; i < nx; i += stride) xbf[i] = f2bf(x[i]);
  for (size_t i = tid; i < (size_t)128 * HID; i += stride) fcWbf[i] = f2bf(fcW[i]);
  for (size_t i = tid; i < (size_t)4 * HID * KIN0; i += stride) wih0bf[i] = f2bf(Wih0[i]);
  for (size_t i = tid; i < (size_t)4 * HID * HID; i += stride) wih1bf[i] = f2bf(Wih1[i]);
  for (size_t i = tid; i < (size_t)4 * HID * HID; i += stride) wih2bf[i] = f2bf(Wih2[i]);
  for (size_t i = tid; i < (size_t)3 * 2 * BATCH * HID; i += stride) hbuf[i] = 0;
  if (tid == 0) *barrier_cnt = 0;
}

// ---------------- persistent pipelined LSTM scan ----------------
extern "C" __global__ void __launch_bounds__(256, 1)
lstm_scan_kernel(const unsigned short* __restrict__ xbf,
                 const unsigned short* __restrict__ wih0bf,
                 const unsigned short* __restrict__ wih1bf,
                 const unsigned short* __restrict__ wih2bf,
                 const float* __restrict__ Whh0, const float* __restrict__ bih0, const float* __restrict__ bhh0,
                 const float* __restrict__ Whh1, const float* __restrict__ bih1, const float* __restrict__ bhh1,
                 const float* __restrict__ Whh2, const float* __restrict__ bih2, const float* __restrict__ bhh2,
                 unsigned short* __restrict__ hbuf,
                 unsigned short* __restrict__ hsout,
                 unsigned* __restrict__ barrier_cnt) {
  extern __shared__ char smem[];
  const int tid = threadIdx.x;
  const int wg = blockIdx.x;
  const int layer = wg / WG_PER_LAYER;
  const int slice = wg % WG_PER_LAYER;
  const int j0 = slice * HSLICE;
  const int KIN = (layer == 0) ? KIN0 : HID;

  unsigned short* whh = (unsigned short*)(smem + LDS_WHH_OFF);    // [NCOLS][HID]
  const unsigned short* xstage = (const unsigned short*)(smem + LDS_XST_OFF);
  const unsigned short* hstage = (const unsigned short*)(smem + LDS_HST_OFF);
  float* gbuf  = (float*)(smem + LDS_GBUF_OFF);                   // [BATCH][NCOLS]
  float* cbuf  = (float*)(smem + LDS_CBUF_OFF);                   // [BATCH][HSLICE]
  float* gbias = (float*)(smem + LDS_BIAS_OFF);                   // [NCOLS]

  const unsigned short* wihbf = (layer == 0) ? wih0bf : (layer == 1) ? wih1bf : wih2bf;
  const float* Whh = (layer == 0) ? Whh0 : (layer == 1) ? Whh1 : Whh2;
  const float* bih = (layer == 0) ? bih0 : (layer == 1) ? bih1 : bih2;
  const float* bhh = (layer == 0) ? bhh0 : (layer == 1) ? bhh1 : bhh2;

  // Stage this WG's Whh slice into LDS (bf16, [col][k]) — once for all 512 steps.
  // Local col c -> global gate row G = (c/HSLICE)*HID + j0 + (c%HSLICE), gates i,f,g,o.
  for (int idx = tid; idx < NCOLS * HID; idx += 256) {
    int c = idx / HID, k = idx - c * HID;
    int G = (c / HSLICE) * HID + j0 + (c % HSLICE);
    whh[c * HID + k] = f2bf(Whh[(size_t)G * HID + k]);
  }
  for (int c = tid; c < NCOLS; c += 256) {
    int G = (c / HSLICE) * HID + j0 + (c % HSLICE);
    gbias[c] = bih[G] + bhh[G];
  }
  for (int e = tid; e < BATCH * HSLICE; e += 256) cbuf[e] = 0.0f;
  __syncthreads();

  const int wave = tid >> 5, lane = tid & 31;
  const int col0 = wave * 16;   // each of 8 waves owns a 16-wide column strip
  // This wave's Wih B-panel in global: columns are contiguous gate rows.
  const unsigned short* wihB =
      wihbf + (size_t)((col0 >> 5) * HID + j0 + (col0 & 31)) * KIN;

  unsigned phase = 0;
  for (int s = 0; s < T_STEPS + 2; ++s) {
    const int t = s - layer;                        // layer-pipelined wavefront
    if (t >= 0 && t < T_STEPS) {
      const unsigned short* xin = (layer == 0)
          ? (xbf + (size_t)t * BATCH * KIN0)
          : (hbuf + (((size_t)(layer - 1) * 2 + (t & 1)) * BATCH * HID));
      const unsigned short* hprev =
          hbuf + (((size_t)layer * 2 + ((t + 1) & 1)) * BATCH * HID);

      // --- async-stage x_t and h_{t-1} into LDS (16B per lane-op, ASYNCcnt) ---
      {
        const int nchx = BATCH * KIN / 8;   // # of 16B chunks (divisible by 256)
        for (int i = tid; i < nchx; i += 256)
          async_copy_b128(LDS_XST_OFF + (unsigned)i * 16u, xin + (size_t)i * 8);
        const int nchh = BATCH * HID / 8;
        for (int i = tid; i < nchh; i += 256)
          async_copy_b128(LDS_HST_OFF + (unsigned)i * 16u, hprev + (size_t)i * 8);
        wait_asynccnt0();
      }
      __syncthreads();

      v8f acc0 = {}, acc1 = {}, acc2 = {}, acc3 = {};
      // x_t @ Wih^T  (K = KIN): A from LDS stage, B streamed from L2 (bf16)
      for (int kc = 0; kc < KIN; kc += 32) {
        v16bf b = load_b_bf16(wihB + kc, KIN, lane);
        v16bf a0 = load_a_bf16(xstage + (size_t)0  * KIN + kc, KIN, lane);
        v16bf a1 = load_a_bf16(xstage + (size_t)16 * KIN + kc, KIN, lane);
        v16bf a2 = load_a_bf16(xstage + (size_t)32 * KIN + kc, KIN, lane);
        v16bf a3 = load_a_bf16(xstage + (size_t)48 * KIN + kc, KIN, lane);
        acc0 = __builtin_amdgcn_wmma_f32_16x16x32_bf16(false, a0, false, b, (short)0, acc0, false, false);
        acc1 = __builtin_amdgcn_wmma_f32_16x16x32_bf16(false, a1, false, b, (short)0, acc1, false, false);
        acc2 = __builtin_amdgcn_wmma_f32_16x16x32_bf16(false, a2, false, b, (short)0, acc2, false, false);
        acc3 = __builtin_amdgcn_wmma_f32_16x16x32_bf16(false, a3, false, b, (short)0, acc3, false, false);
      }
      // h_{t-1} @ Whh^T  (K = 512): A + B both from LDS
      for (int kc = 0; kc < HID; kc += 32) {
        v16bf b = load_b_bf16(whh + (size_t)col0 * HID + kc, HID, lane);
        v16bf a0 = load_a_bf16(hstage + (size_t)0  * HID + kc, HID, lane);
        v16bf a1 = load_a_bf16(hstage + (size_t)16 * HID + kc, HID, lane);
        v16bf a2 = load_a_bf16(hstage + (size_t)32 * HID + kc, HID, lane);
        v16bf a3 = load_a_bf16(hstage + (size_t)48 * HID + kc, HID, lane);
        acc0 = __builtin_amdgcn_wmma_f32_16x16x32_bf16(false, a0, false, b, (short)0, acc0, false, false);
        acc1 = __builtin_amdgcn_wmma_f32_16x16x32_bf16(false, a1, false, b, (short)0, acc1, false, false);
        acc2 = __builtin_amdgcn_wmma_f32_16x16x32_bf16(false, a2, false, b, (short)0, acc2, false, false);
        acc3 = __builtin_amdgcn_wmma_f32_16x16x32_bf16(false, a3, false, b, (short)0, acc3, false, false);
      }
      // C layout: VGPR r -> row (r + 8*(lane>>4)), col = lane&15
      {
        const int n = col0 + (lane & 15);
        const int mb = (lane >> 4) * 8;
        const float bb = gbias[n];
#pragma unroll
        for (int r = 0; r < 8; ++r) {
          gbuf[(0 * 16 + mb + r) * NCOLS + n] = acc0[r] + bb;
          gbuf[(1 * 16 + mb + r) * NCOLS + n] = acc1[r] + bb;
          gbuf[(2 * 16 + mb + r) * NCOLS + n] = acc2[r] + bb;
          gbuf[(3 * 16 + mb + r) * NCOLS + n] = acc3[r] + bb;
        }
      }
      __syncthreads();
      // Gate fusion: all four gates of this WG's hidden units are local.
      unsigned short* hout = hbuf + (((size_t)layer * 2 + (t & 1)) * BATCH * HID);
      for (int e = tid; e < BATCH * HSLICE; e += 256) {
        const int row = e / HSLICE, jj = e - row * HSLICE;
        const float gi = gbuf[row * NCOLS + 0 * HSLICE + jj];
        const float gf = gbuf[row * NCOLS + 1 * HSLICE + jj];
        const float gg = gbuf[row * NCOLS + 2 * HSLICE + jj];
        const float go = gbuf[row * NCOLS + 3 * HSLICE + jj];
        const float c = sigmoidf_(gf) * cbuf[e] + sigmoidf_(gi) * tanhf(gg);
        cbuf[e] = c;
        const float h = sigmoidf_(go) * tanhf(c);
        const unsigned short hb = f2bf(h);
        hout[(size_t)row * HID + j0 + jj] = hb;
        if (layer == 2 && t >= T_STEPS - CHUNK)
          hsout[((size_t)(t - (T_STEPS - CHUNK)) * BATCH + row) * HID + j0 + jj] = hb;
      }
    }
    // device-wide barrier between macro-steps
    __syncthreads();
    ++phase;
    if (tid == 0) {
      __threadfence();
      __hip_atomic_fetch_add(barrier_cnt, 1u, __ATOMIC_RELEASE, __HIP_MEMORY_SCOPE_AGENT);
      const unsigned target = phase * (unsigned)NWG_LSTM;
      while (__hip_atomic_load(barrier_cnt, __ATOMIC_ACQUIRE, __HIP_MEMORY_SCOPE_AGENT) < target)
        __builtin_amdgcn_s_sleep(2);
    }
    __syncthreads();
  }
}

// ---------------- final FC: (4096 x 512) @ (512 x 128) + b ----------------
extern "C" __global__ void __launch_bounds__(256)
lstm_fc_kernel(const unsigned short* __restrict__ hsout,   // [4096][512] bf16
               const unsigned short* __restrict__ fcWbf,   // [128][512] bf16 ([n][k])
               const float* __restrict__ fcb,
               float* __restrict__ out) {                  // [4096][128] f32
  const int tid = threadIdx.x;
  const int wave = tid >> 5, lane = tid & 31;
  const int m0 = blockIdx.x * 128 + wave * 16;

  v8f acc[8] = {};
  for (int kc = 0; kc < HID; kc += 32) {
    v16bf a = load_a_bf16(hsout + (size_t)m0 * HID + kc, HID, lane);
#pragma unroll
    for (int nt = 0; nt < 8; ++nt) {
      v16bf b = load_b_bf16(fcWbf + (size_t)(nt * 16) * HID + kc, HID, lane);
      acc[nt] = __builtin_amdgcn_wmma_f32_16x16x32_bf16(false, a, false, b, (short)0, acc[nt], false, false);
    }
  }
  const int mb = (lane >> 4) * 8;
  const int nc = lane & 15;
#pragma unroll
  for (int nt = 0; nt < 8; ++nt) {
    const int n = nt * 16 + nc;
    const float bb = fcb[n];
#pragma unroll
    for (int r = 0; r < 8; ++r)
      out[(size_t)(m0 + mb + r) * 128 + n] = acc[nt][r] + bb;
  }
}

extern "C" void kernel_launch(void* const* d_in, const int* in_sizes, int n_in,
                              void* d_out, int out_size, void* d_ws, size_t ws_size,
                              hipStream_t stream) {
  const float* x     = (const float*)d_in[0];
  const float* Wih0  = (const float*)d_in[1];
  const float* Whh0  = (const float*)d_in[2];
  const float* bih0  = (const float*)d_in[3];
  const float* bhh0  = (const float*)d_in[4];
  const float* Wih1  = (const float*)d_in[5];
  const float* Whh1  = (const float*)d_in[6];
  const float* bih1  = (const float*)d_in[7];
  const float* bhh1  = (const float*)d_in[8];
  const float* Wih2  = (const float*)d_in[9];
  const float* Whh2  = (const float*)d_in[10];
  const float* bih2  = (const float*)d_in[11];
  const float* bhh2  = (const float*)d_in[12];
  const float* fcW   = (const float*)d_in[13];
  const float* fcb   = (const float*)d_in[14];

  char* ws = (char*)d_ws;
  unsigned short* xbf    = (unsigned short*)(ws + WS_XBF_OFF);
  unsigned short* fcWbf  = (unsigned short*)(ws + WS_FCW_OFF);
  unsigned short* hbuf   = (unsigned short*)(ws + WS_HBUF_OFF);
  unsigned short* hsout  = (unsigned short*)(ws + WS_HS_OFF);
  unsigned short* wih0bf = (unsigned short*)(ws + WS_WIH0_OFF);
  unsigned short* wih1bf = (unsigned short*)(ws + WS_WIH1_OFF);
  unsigned short* wih2bf = (unsigned short*)(ws + WS_WIH2_OFF);
  unsigned*       bar    = (unsigned*)(ws + WS_BAR_OFF);

  lstm_prep_kernel<<<1024, 256, 0, stream>>>(x, fcW, Wih0, Wih1, Wih2,
                                             xbf, fcWbf, wih0bf, wih1bf, wih2bf,
                                             hbuf, bar);

  lstm_scan_kernel<<<NWG_LSTM, 256, LDS_TOTAL, stream>>>(
      xbf, wih0bf, wih1bf, wih2bf,
      Whh0, bih0, bhh0, Whh1, bih1, bhh1, Whh2, bih2, bhh2,
      hbuf, hsout, bar);

  lstm_fc_kernel<<<32, 256, 0, stream>>>(hsout, fcWbf, fcb, (float*)d_out);
}
